// QuantizedHI4B1CLinear_26912265077060
// MI455X (gfx1250) — compile-verified, matrix operand order checked by reading, not run on hardware
//
#include <hip/hip_runtime.h>
#include <hip/hip_bf16.h>

typedef __attribute__((ext_vector_type(16))) _Float16 v16h;
typedef __attribute__((ext_vector_type(8)))  _Float16 v8h;
typedef __attribute__((ext_vector_type(8)))  float    v8f;
typedef __attribute__((ext_vector_type(4)))  int      v4i;

#define B_ROWS 8192
#define N_DIM  4096   // K dimension / input feature dim
#define M_DIM  4096   // output feature dim

#define AS1 __attribute__((address_space(1)))
#define AS3 __attribute__((address_space(3)))

#if defined(__has_builtin)
#if __has_builtin(__builtin_amdgcn_global_load_async_to_lds_b128)
#define HAVE_ASYNC_LDS 1
#endif
#endif

// Build address-space pointers from integers (avoids generic->AS addrspacecast):
// generic LDS pointers carry the LDS byte offset in their low 32 bits;
// generic global pointers are the plain 64-bit VA.
static __device__ __forceinline__ AS1 v4i* glob_v4(const void* p) {
    return (AS1 v4i*)(unsigned long long)p;
}
static __device__ __forceinline__ AS3 v4i* lds_v4(const void* p) {
    return (AS3 v4i*)(unsigned int)(unsigned long long)p;
}

static __device__ __forceinline__ void wait_async0() {
#if defined(__has_builtin) && __has_builtin(__builtin_amdgcn_s_wait_asynccnt)
    __builtin_amdgcn_s_wait_asynccnt(0);
#else
    asm volatile("s_wait_asynccnt 0x0" ::: "memory");
#endif
}

// ---------------------------------------------------------------------------
// Kernel 1: decode 4-bit codebook indices -> fp16 weights  W[m][k] = q - 7.5
// ---------------------------------------------------------------------------
__global__ void __launch_bounds__(256) decode_weights(const int* __restrict__ q,
                                                      _Float16* __restrict__ w) {
    const int base = (blockIdx.x * 256 + threadIdx.x) * 8;
    #pragma unroll
    for (int j = 0; j < 8; ++j) {
        w[base + j] = (_Float16)((float)q[base + j] - 7.5f);
    }
}

// ---------------------------------------------------------------------------
// Kernel 2: per-row x = WHT(input * SU) * (1/65536), cast fp16.
// ---------------------------------------------------------------------------
__global__ void __launch_bounds__(256) had_in(const float* __restrict__ x,
                                              const float* __restrict__ su,
                                              _Float16* __restrict__ xh) {
    __shared__ float lds[N_DIM];
    const int row = blockIdx.x;
    const int tid = threadIdx.x;
    const float* rp = x + (size_t)row * N_DIM;

    #pragma unroll
    for (int j = 0; j < N_DIM / 256; ++j) {
        const int idx = tid + j * 256;
        lds[idx] = rp[idx] * su[idx];
    }
    __syncthreads();

    #pragma unroll 1
    for (int s = 0; s < 12; ++s) {
        const int h = 1 << s;
        #pragma unroll
        for (int j = 0; j < (N_DIM / 2) / 256; ++j) {
            const int p = tid + j * 256;
            const int i = ((p >> s) << (s + 1)) + (p & (h - 1));
            const float a = lds[i];
            const float b = lds[i + h];
            lds[i]     = a + b;
            lds[i + h] = a - b;
        }
        __syncthreads();
    }

    const float scale = 1.0f / 65536.0f;   // 1/sqrt(4096) * 1/1024
    _Float16* op = xh + (size_t)row * N_DIM;
    #pragma unroll
    for (int j = 0; j < N_DIM / 256; ++j) {
        const int idx = tid + j * 256;
        op[idx] = (_Float16)(lds[idx] * scale);
    }
}

// ---------------------------------------------------------------------------
// Kernel 3: z = Xh @ Wh^T via fp16 WMMA, fp32 accum, scaled by Wscale*1024.
//
// Block = 256 threads (8 waves), block tile 128x128, BK=32.
// Wave grid 2(M) x 4(N); each wave computes 64x32 via 4x2 = 8 WMMA tiles.
// A/B staged to LDS with gfx1250 async-to-LDS loads when available.
//
// Fragment register layout per CDNA5 ISA (16-bit A 16x32 / mirrored B 32x16):
//   lane L: m (or n) = L & 15, K-base = (L>>4)*8
//   halves 0..7  = K base..base+7      (ds_load_b128)
//   halves 8..15 = K base+16..base+23  (ds_load_b128)
// ---------------------------------------------------------------------------
__global__ void __launch_bounds__(256) gemm_wmma(const _Float16* __restrict__ Xh,
                                                 const _Float16* __restrict__ Wh,
                                                 const float* __restrict__ wscale,
                                                 float* __restrict__ out) {
    __shared__ _Float16 sA[128 * 32];
    __shared__ _Float16 sB[128 * 32];

    const int tid  = threadIdx.x;
    const int lane = tid & 31;
    const int wave = tid >> 5;
    const int row0 = blockIdx.x * 128;
    const int col0 = blockIdx.y * 128;
    const int wr0  = (wave >> 2) * 64;   // wave M offset within block tile
    const int wc0  = (wave & 3) * 32;    // wave N offset within block tile
    const int r    = lane & 15;          // M index (A) / N index (B)
    const int kg   = lane >> 4;          // K sub-block select (0 or 1)

    v8f acc[4][2] = {};

    #pragma unroll 1
    for (int k0 = 0; k0 < N_DIM; k0 += 32) {
        // ---- stage A(128x32) and B(128x32) tiles into LDS ----
        #pragma unroll
        for (int j = 0; j < 2; ++j) {
            const int c    = j * 256 + tid;      // 16-byte chunk id, 0..511
            const int rrow = c >> 2;             // tile row 0..127
            const int seg  = (c & 3) * 8;        // half-offset within row
            const _Float16* ga = Xh + (size_t)(row0 + rrow) * N_DIM + k0 + seg;
            const _Float16* gb = Wh + (size_t)(col0 + rrow) * N_DIM + k0 + seg;
            const int loff = rrow * 32 + seg;
#ifdef HAVE_ASYNC_LDS
            __builtin_amdgcn_global_load_async_to_lds_b128(
                glob_v4(ga), lds_v4(&sA[loff]), 0, 0);
            __builtin_amdgcn_global_load_async_to_lds_b128(
                glob_v4(gb), lds_v4(&sB[loff]), 0, 0);
#else
            *(v8h*)&sA[loff] = *(const v8h*)ga;
            *(v8h*)&sB[loff] = *(const v8h*)gb;
#endif
        }
#ifdef HAVE_ASYNC_LDS
        wait_async0();
#endif
        __syncthreads();

        // ---- load fragments from LDS and issue 8 WMMAs ----
        union { v16h v; v8h h[2]; } Afrag[4], Bfrag[2];
        #pragma unroll
        for (int mt = 0; mt < 4; ++mt) {
            const int base = (wr0 + mt * 16 + r) * 32 + kg * 8;
            Afrag[mt].h[0] = *(const v8h*)&sA[base];
            Afrag[mt].h[1] = *(const v8h*)&sA[base + 16];
        }
        #pragma unroll
        for (int nt = 0; nt < 2; ++nt) {
            const int base = (wc0 + nt * 16 + r) * 32 + kg * 8;
            Bfrag[nt].h[0] = *(const v8h*)&sB[base];
            Bfrag[nt].h[1] = *(const v8h*)&sB[base + 16];
        }
        #pragma unroll
        for (int mt = 0; mt < 4; ++mt) {
            #pragma unroll
            for (int nt = 0; nt < 2; ++nt) {
                acc[mt][nt] = __builtin_amdgcn_wmma_f32_16x16x32_f16(
                    false, Afrag[mt].v, false, Bfrag[nt].v,
                    (short)0, acc[mt][nt], false, false);
            }
        }
        __syncthreads();
    }

    // ---- epilogue: scale and store ----
    const float s = wscale[0] * 1024.0f;
    #pragma unroll
    for (int mt = 0; mt < 4; ++mt) {
        #pragma unroll
        for (int nt = 0; nt < 2; ++nt) {
            const int gc = col0 + wc0 + nt * 16 + r;
            #pragma unroll
            for (int v = 0; v < 8; ++v) {
                const int gr = row0 + wr0 + mt * 16 + v + kg * 8;
                out[(size_t)gr * M_DIM + gc] = acc[mt][nt][v] * s;
            }
        }
    }
}

// ---------------------------------------------------------------------------
// Kernel 4: in-place per-row out = WHT(z) * (1/64) * SV
// ---------------------------------------------------------------------------
__global__ void __launch_bounds__(256) had_out(float* __restrict__ z,
                                               const float* __restrict__ sv) {
    __shared__ float lds[N_DIM];
    const int row = blockIdx.x;
    const int tid = threadIdx.x;
    float* rp = z + (size_t)row * M_DIM;

    #pragma unroll
    for (int j = 0; j < M_DIM / 256; ++j) {
        const int idx = tid + j * 256;
        lds[idx] = rp[idx];
    }
    __syncthreads();

    #pragma unroll 1
    for (int s = 0; s < 12; ++s) {
        const int h = 1 << s;
        #pragma unroll
        for (int j = 0; j < (M_DIM / 2) / 256; ++j) {
            const int p = tid + j * 256;
            const int i = ((p >> s) << (s + 1)) + (p & (h - 1));
            const float a = lds[i];
            const float b = lds[i + h];
            lds[i]     = a + b;
            lds[i + h] = a - b;
        }
        __syncthreads();
    }

    const float scale = 1.0f / 64.0f;   // 1/sqrt(4096)
    #pragma unroll
    for (int j = 0; j < M_DIM / 256; ++j) {
        const int idx = tid + j * 256;
        rp[idx] = lds[idx] * scale * sv[idx];
    }
}

// ---------------------------------------------------------------------------
// Host-side launcher
// Inputs: 0 input (B,N) f32 | 1 Qidxs (M,N) i32 | 2 SU f32 | 3 SV f32
//         4 Wscale f32 | 5..8 trivial Hadamard factors (unused, K=1)
// Workspace: [0, 64MB) Xh fp16 activations; [64MB, 96MB) Wh fp16 weights
// ---------------------------------------------------------------------------
extern "C" void kernel_launch(void* const* d_in, const int* in_sizes, int n_in,
                              void* d_out, int out_size, void* d_ws, size_t ws_size,
                              hipStream_t stream) {
    const float* input  = (const float*)d_in[0];
    const int*   qidxs  = (const int*)d_in[1];
    const float* su     = (const float*)d_in[2];
    const float* sv     = (const float*)d_in[3];
    const float* wscale = (const float*)d_in[4];

    _Float16* Xh = (_Float16*)d_ws;
    _Float16* Wh = (_Float16*)((char*)d_ws + (size_t)B_ROWS * N_DIM * sizeof(_Float16));
    float*    out = (float*)d_out;

    decode_weights<<<dim3((M_DIM * N_DIM) / (256 * 8)), 256, 0, stream>>>(qidxs, Wh);
    had_in<<<dim3(B_ROWS), 256, 0, stream>>>(input, su, Xh);
    gemm_wmma<<<dim3(B_ROWS / 128, M_DIM / 128), 256, 0, stream>>>(Xh, Wh, wscale, out);
    had_out<<<dim3(B_ROWS), 256, 0, stream>>>(out, sv);
}
